// RPNHead_13692355740311
// MI455X (gfx1250) — compile-verified, matrix operand order checked by reading, not run on hardware
//
#include <hip/hip_runtime.h>
#include <stdint.h>

#define B_      2
#define C_      256
#define H_      200
#define W_      304
#define HP_     202              // padded H (1px halo)
#define WP_     306              // padded W (1px halo)
#define A_      3
#define HW_     (H_*W_)          // 60800
#define NPOS_   (B_*HW_)         // 121600
#define SPI_    (HW_*A_)         // 182400 scores per image
#define NSEL_   2000
#define NBUCKET_ 4096
#define CANDCAP_ 4096
#define MAXR_   4.135166556742356f
#define IOU_THR_ 0.7f

typedef __attribute__((ext_vector_type(16))) __bf16   bf16x16;
typedef __attribute__((ext_vector_type(8)))  float    f32x8;
typedef __attribute__((ext_vector_type(4)))  uint32_t u32x4;

union ABu { bf16x16 v; u32x4 u[2]; };

static __device__ __forceinline__ uint32_t f2bf_bits(float f) {
  uint32_t u = __float_as_uint(f);
  u += 0x7FFFu + ((u >> 16) & 1u);        // round-to-nearest-even
  return u >> 16;
}
static __device__ __forceinline__ uint32_t sortkey(float f) {
  uint32_t u = __float_as_uint(f);
  return (u & 0x80000000u) ? ~u : (u | 0x80000000u);
}

// ---------------- init: zero histograms + counters -------------------------
__global__ void k_init(unsigned* hist) {
  int i = blockIdx.x * blockDim.x + threadIdx.x;
  if (i < 2*NBUCKET_ + 4) hist[i] = 0u;
}

// ---------------- NCHW f32 -> padded NHWC bf16 (coalesced both sides) ------
// thread = (b, c8, h, w): 8 coalesced f32 reads, one contiguous b128 store
__global__ __launch_bounds__(256) void k_convert_x(const float* __restrict__ x,
                                                   uint16_t* __restrict__ xp) {
  size_t i = (size_t)blockIdx.x * blockDim.x + threadIdx.x;
  if (i >= (size_t)B_*32*H_*W_) return;
  const int w  = (int)(i % W_);
  size_t t = i / W_;
  const int h  = (int)(t % H_); t /= H_;
  const int c8 = (int)(t % 32);
  const int b  = (int)(t / 32);
  const float* src = x + ((size_t)(b*C_ + c8*8)*H_ + h)*W_ + w;
  uint32_t p[4];
  #pragma unroll
  for (int j = 0; j < 4; ++j) {
    const uint32_t lo = f2bf_bits(src[(size_t)(2*j  )*HW_]);
    const uint32_t hi = f2bf_bits(src[(size_t)(2*j+1)*HW_]);
    p[j] = lo | (hi << 16);
  }
  u32x4 v = {p[0], p[1], p[2], p[3]};
  *reinterpret_cast<u32x4*>(xp + ((size_t)(b*HP_ + h+1)*WP_ + (w+1))*C_ + c8*8) = v;
}

// ---------------- zero the 1px halo of the padded tensor -------------------
__global__ void k_halo(uint16_t* __restrict__ xp) {
  int i = blockIdx.x * blockDim.x + threadIdx.x;
  if (i >= B_*1012*32) return;              // 1012 border cells/image, 32 c-groups
  const int c8 = i % 32;
  int q = (i / 32) % 1012;
  const int b = i / (32*1012);
  int h, w;
  if      (q < 306) { h = 0;         w = q;       }
  else if (q < 612) { h = HP_-1;     w = q - 306; }
  else if (q < 812) { h = q - 612+1; w = 0;       }
  else              { h = q - 812+1; w = WP_-1;   }
  u32x4 z = {0u,0u,0u,0u};
  *reinterpret_cast<u32x4*>(xp + ((size_t)(b*HP_ + h)*WP_ + w)*C_ + c8*8) = z;
}

// ---------------- pack 3x3 conv weights into B-fragment layout -------------
// wpack[t][nt][kc][lane][e] : n = nt*16 + (lane&15); khalf = (lane>>4)*8;
//                            k = kc*32 + (e&8 ? 16:0) + khalf + (e&7)
__global__ __launch_bounds__(256) void k_pack_w(const float* __restrict__ cw,
                                                uint16_t* __restrict__ wpack) {
  int i = blockIdx.x * blockDim.x + threadIdx.x;
  if (i >= 9*16*8*32*16) return;
  int e    = i & 15;
  int lane = (i >> 4) & 31;
  int kc   = (i >> 9) & 7;
  int nt   = (i >> 12) & 15;
  int t    = i >> 16;
  int n = nt*16 + (lane & 15);
  int k = kc*32 + ((e & 8) ? 16 : 0) + ((lane >> 4) << 3) + (e & 7);
  wpack[i] = (uint16_t)f2bf_bits(cw[(size_t)n*2304 + (size_t)k*9 + t]);
}

// ---------------- pack fused head weights (cls 0..2, reg 3..14, pad 15) ----
__global__ void k_pack_hw(const float* __restrict__ clsw,
                          const float* __restrict__ regw,
                          uint16_t* __restrict__ hwp) {
  int i = blockIdx.x * blockDim.x + threadIdx.x;
  if (i >= 8*32*16) return;
  int e    = i & 15;
  int lane = (i >> 4) & 31;
  int kc   = i >> 9;
  int n = lane & 15;
  int k = kc*32 + ((e & 8) ? 16 : 0) + ((lane >> 4) << 3) + (e & 7);
  float v = 0.f;
  if (n < 3)       v = clsw[(size_t)n*C_ + k];
  else if (n < 15) v = regw[(size_t)(n-3)*C_ + k];
  hwp[i] = (uint16_t)f2bf_bits(v);
}

__global__ void k_pack_hbias(const float* __restrict__ clsb,
                             const float* __restrict__ regb,
                             float* __restrict__ hb) {
  int n = threadIdx.x;
  if (n >= 16) return;
  hb[n] = (n < 3) ? clsb[n] : ((n < 15) ? regb[n-3] : 0.f);
}

// ---------------- 3x3 conv via bf16 WMMA (implicit GEMM, halo-padded) ------
// wave tile: M=16 w-positions x N=32 out-channels (2 N-tiles), K = 9 taps x 256
__global__ __launch_bounds__(256) void k_conv(const uint16_t* __restrict__ xp,
                                              const uint16_t* __restrict__ wpack,
                                              const float* __restrict__ convb,
                                              uint16_t* __restrict__ feat) {
  const int lane = threadIdx.x & 31;
  const int wid  = blockIdx.x * 8 + (threadIdx.x >> 5);
  const int ntp = wid & 7;                 // pair of N-tiles: nt0=2*ntp, nt1=2*ntp+1
  int rest = wid >> 3;
  const int wt = rest % 19;
  const int bh = rest / 19;                // b*H + h
  const int h  = bh % H_;
  const int b  = bh / H_;
  const int w0 = wt * 16;
  const int mrow  = lane & 15;
  const int khalf = (lane >> 4) << 3;      // 0 or 8

  f32x8 acc0 = {0.f,0.f,0.f,0.f,0.f,0.f,0.f,0.f};
  f32x8 acc1 = {0.f,0.f,0.f,0.f,0.f,0.f,0.f,0.f};

  for (int t = 0; t < 9; ++t) {
    const int dy = t / 3, dx = t - dy*3;
    // padded coords: (h+dy, w0+mrow+dx) always in range
    const uint16_t* ap  = xp + ((size_t)(b*HP_ + h + dy)*WP_ + (w0 + mrow + dx))*C_;
    const uint16_t* wp0 = wpack + ((((size_t)t*16 + 2*ntp    )*8)*32 + lane)*16;
    const uint16_t* wp1 = wpack + ((((size_t)t*16 + 2*ntp + 1)*8)*32 + lane)*16;
    #pragma unroll
    for (int kc = 0; kc < 8; ++kc) {
      ABu av, bv0, bv1;
      const uint16_t* p = ap + kc*32 + khalf;
      av.u[0] = *reinterpret_cast<const u32x4*>(p);
      av.u[1] = *reinterpret_cast<const u32x4*>(p + 16);
      const uint16_t* q0 = wp0 + (size_t)kc*32*16;
      bv0.u[0] = *reinterpret_cast<const u32x4*>(q0);
      bv0.u[1] = *reinterpret_cast<const u32x4*>(q0 + 8);
      const uint16_t* q1 = wp1 + (size_t)kc*32*16;
      bv1.u[0] = *reinterpret_cast<const u32x4*>(q1);
      bv1.u[1] = *reinterpret_cast<const u32x4*>(q1 + 8);
      acc0 = __builtin_amdgcn_wmma_f32_16x16x32_bf16(false, av.v, false, bv0.v,
                                                     (short)0, acc0, false, false);
      acc1 = __builtin_amdgcn_wmma_f32_16x16x32_bf16(false, av.v, false, bv1.v,
                                                     (short)0, acc1, false, false);
    }
  }
  // D layout: VGPR v -> M = v + (lane>=16 ? 8 : 0), N = lane&15
  const int mbase = (lane >> 4) << 3;
  const int n0 = (2*ntp    )*16 + (lane & 15);
  const int n1 = (2*ntp + 1)*16 + (lane & 15);
  const float bias0 = convb[n0];
  const float bias1 = convb[n1];
  #pragma unroll
  for (int v = 0; v < 8; ++v) {
    const int m = v + mbase;
    const size_t rowoff = ((size_t)(b*H_ + h)*W_ + (w0 + m))*C_;
    float v0 = acc0[v] + bias0; v0 = v0 > 0.f ? v0 : 0.f;
    float v1 = acc1[v] + bias1; v1 = v1 > 0.f ? v1 : 0.f;
    feat[rowoff + n0] = (uint16_t)f2bf_bits(v0);
    feat[rowoff + n1] = (uint16_t)f2bf_bits(v1);
  }
}

// ---------------- fused 1x1 head via WMMA (N=16: 3 cls + 12 reg + pad) -----
__global__ __launch_bounds__(256) void k_head(const uint16_t* __restrict__ feat,
                                              const uint16_t* __restrict__ hwp,
                                              const float* __restrict__ hbias,
                                              float* __restrict__ hout) {
  const int lane = threadIdx.x & 31;
  const int wid  = blockIdx.x * 8 + (threadIdx.x >> 5);
  const int p0 = wid * 16;
  const int mrow  = lane & 15;
  const int khalf = (lane >> 4) << 3;
  const uint16_t* ap = feat + (size_t)(p0 + mrow)*C_;

  f32x8 acc = {0.f,0.f,0.f,0.f,0.f,0.f,0.f,0.f};
  #pragma unroll
  for (int kc = 0; kc < 8; ++kc) {
    ABu av, bv;
    const uint16_t* p = ap + kc*32 + khalf;
    av.u[0] = *reinterpret_cast<const u32x4*>(p);
    av.u[1] = *reinterpret_cast<const u32x4*>(p + 16);
    const uint16_t* wp = hwp + ((size_t)kc*32 + lane)*16;
    bv.u[0] = *reinterpret_cast<const u32x4*>(wp);
    bv.u[1] = *reinterpret_cast<const u32x4*>(wp + 8);
    acc = __builtin_amdgcn_wmma_f32_16x16x32_bf16(false, av.v, false, bv.v,
                                                  (short)0, acc, false, false);
  }
  const int n = lane & 15;
  const float bias = hbias[n];
  const int mbase = (lane >> 4) << 3;
  #pragma unroll
  for (int v = 0; v < 8; ++v) {
    const int pos = p0 + v + mbase;
    hout[(size_t)pos*16 + n] = acc[v] + bias;
  }
}

// ---------------- score histogram (top-12-bit buckets of sortable key) -----
__global__ __launch_bounds__(256) void k_hist(const float* __restrict__ hout,
                                              unsigned* __restrict__ hist) {
  int i = blockIdx.x * blockDim.x + threadIdx.x;
  if (i >= B_*SPI_) return;
  const int b = i / SPI_;
  const int s = i - b*SPI_;
  const int pos = s / 3, a = s - pos*3;
  const float lg = hout[(size_t)(b*HW_ + pos)*16 + a];
  atomicAdd(&hist[b*NBUCKET_ + (sortkey(lg) >> 20)], 1u);
}

// ---------------- find threshold bucket ------------------------------------
__global__ void k_thresh(const unsigned* __restrict__ hist, int* __restrict__ thrB) {
  if (threadIdx.x != 0) return;
  const unsigned* hb = hist + blockIdx.x * NBUCKET_;
  unsigned cum = 0;
  int T = 0;
  for (int bk = NBUCKET_-1; bk >= 0; --bk) {
    unsigned c = hb[bk];
    if (cum + c >= (unsigned)NSEL_) { T = bk; break; }
    cum += c;
  }
  thrB[blockIdx.x] = T;
}

// ---------------- collect candidates (bucket >= T) with unique 64-bit keys -
__global__ __launch_bounds__(256) void k_collect(const float* __restrict__ hout,
                                                 const int* __restrict__ thrB,
                                                 unsigned* __restrict__ cnt,
                                                 unsigned long long* __restrict__ cand) {
  int i = blockIdx.x * blockDim.x + threadIdx.x;
  if (i >= B_*SPI_) return;
  const int b = i / SPI_;
  const int s = i - b*SPI_;
  const int pos = s / 3, a = s - pos*3;
  const float lg = hout[(size_t)(b*HW_ + pos)*16 + a];
  const uint32_t k32 = sortkey(lg);
  if ((int)(k32 >> 20) >= thrB[b]) {
    unsigned p = atomicAdd(&cnt[b], 1u);
    if (p < CANDCAP_)
      cand[(size_t)b*CANDCAP_ + p] =
        ((unsigned long long)k32 << 32) | (unsigned long long)(0xFFFFFFFFu - (uint32_t)s);
  }
}

// ---------------- per-image bitonic sort (descending, unique keys) ---------
__global__ __launch_bounds__(1024) void k_sort(const unsigned long long* __restrict__ cand,
                                               const unsigned* __restrict__ cnt,
                                               float* __restrict__ topkL,
                                               int* __restrict__ topkI) {
  __shared__ unsigned long long sk[CANDCAP_];
  const int b = blockIdx.x, tid = threadIdx.x;
  const int n = (int)min(cnt[b], (unsigned)CANDCAP_);
  for (int l = tid; l < CANDCAP_; l += 1024)
    sk[l] = (l < n) ? cand[(size_t)b*CANDCAP_ + l] : 0ull;
  __syncthreads();
  for (int k = 2; k <= CANDCAP_; k <<= 1) {
    for (int j = k >> 1; j > 0; j >>= 1) {
      for (int l = tid; l < CANDCAP_; l += 1024) {
        const int ixj = l ^ j;
        if (ixj > l) {
          const unsigned long long a0 = sk[l], a1 = sk[ixj];
          const bool up = ((l & k) == 0);
          if (up ? (a0 < a1) : (a0 > a1)) { sk[l] = a1; sk[ixj] = a0; }
        }
      }
      __syncthreads();
    }
  }
  for (int l = tid; l < NSEL_; l += 1024) {
    const unsigned long long kk = sk[l];
    const uint32_t k32 = (uint32_t)(kk >> 32);
    const uint32_t u = (k32 & 0x80000000u) ? (k32 ^ 0x80000000u) : ~k32;
    topkL[b*NSEL_ + l] = __uint_as_float(u);
    topkI[b*NSEL_ + l] = (int)(0xFFFFFFFFu - (uint32_t)kk);
  }
}

// ---------------- decode boxes (anchors computed analytically) -------------
__global__ __launch_bounds__(256) void k_decode(const float* __restrict__ hout,
                                                const float* __restrict__ topkL,
                                                const int* __restrict__ topkI,
                                                float* __restrict__ boxes,
                                                float* __restrict__ out) {
  int i = blockIdx.x * blockDim.x + threadIdx.x;
  if (i >= B_*NSEL_) return;
  const int b = i / NSEL_;
  const float lg = topkL[i];
  const int idx = topkI[i];
  const int pos = idx / 3, a = idx - pos*3;
  const int hh = pos / W_, ww = pos - hh*W_;
  // rounded base anchors for stride 4, scale 8, ratios {0.5,1,2}
  const float bx1[3] = {-21.f, -14.f,  -9.f};
  const float by1[3] = { -9.f, -14.f, -21.f};
  const float bx2[3] = { 24.f,  17.f,  12.f};
  const float by2[3] = { 12.f,  17.f,  24.f};
  const float ax1 = bx1[a] + 4.f*ww, ay1 = by1[a] + 4.f*hh;
  const float ax2 = bx2[a] + 4.f*ww, ay2 = by2[a] + 4.f*hh;
  const float* d = hout + (size_t)(b*HW_ + pos)*16 + 3 + 4*a;
  const float dx = d[0], dy = d[1];
  const float dw = fminf(fmaxf(d[2], -MAXR_), MAXR_);
  const float dh = fminf(fmaxf(d[3], -MAXR_), MAXR_);
  const float px = (ax1 + ax2)*0.5f, py = (ay1 + ay2)*0.5f;
  const float pw = ax2 - ax1 + 1.f,  ph = ay2 - ay1 + 1.f;
  const float gw = pw * expf(dw), gh = ph * expf(dh);
  const float gx = px + pw*dx,    gy = py + ph*dy;
  const float x1 = fminf(fmaxf(gx - gw*0.5f + 0.5f, 0.f), 1215.f);
  const float y1 = fminf(fmaxf(gy - gh*0.5f + 0.5f, 0.f),  799.f);
  const float x2 = fminf(fmaxf(gx + gw*0.5f - 0.5f, 0.f), 1215.f);
  const float y2 = fminf(fmaxf(gy + gh*0.5f - 0.5f, 0.f),  799.f);
  const float sc = 1.f / (1.f + expf(-lg));
  float* bp = boxes + (size_t)i*4;
  bp[0] = x1; bp[1] = y1; bp[2] = x2; bp[3] = y2;
  float* op = out + (size_t)i*5;
  op[0] = x1; op[1] = y1; op[2] = x2; op[3] = y2; op[4] = sc;
}

// ---------------- greedy NMS, one block per image --------------------------
__global__ __launch_bounds__(1024) void k_nms(const float* __restrict__ boxes,
                                              float* __restrict__ out) {
  __shared__ float sx1[NSEL_], sy1[NSEL_], sx2[NSEL_], sy2[NSEL_], sar[NSEL_];
  __shared__ int skp[NSEL_];
  const int b = blockIdx.x, tid = threadIdx.x;
  for (int j = tid; j < NSEL_; j += 1024) {
    const float* bp = boxes + (size_t)(b*NSEL_ + j)*4;
    sx1[j] = bp[0]; sy1[j] = bp[1]; sx2[j] = bp[2]; sy2[j] = bp[3];
    sar[j] = (bp[2]-bp[0]+1.f)*(bp[3]-bp[1]+1.f);
    skp[j] = 1;
  }
  __syncthreads();
  for (int i = 0; i < NSEL_; ++i) {
    if (skp[i]) {
      const float ix1 = sx1[i], iy1 = sy1[i], ix2 = sx2[i], iy2 = sy2[i], ia = sar[i];
      for (int j = tid; j < NSEL_; j += 1024) {
        if (j > i && skp[j]) {
          const float xx1 = fmaxf(ix1, sx1[j]);
          const float yy1 = fmaxf(iy1, sy1[j]);
          const float xx2 = fminf(ix2, sx2[j]);
          const float yy2 = fminf(iy2, sy2[j]);
          const float inter = fmaxf(0.f, xx2-xx1+1.f) * fmaxf(0.f, yy2-yy1+1.f);
          if (inter / (ia + sar[j] - inter) > IOU_THR_) skp[j] = 0;
        }
      }
    }
    __syncthreads();
  }
  for (int j = tid; j < NSEL_; j += 1024) {
    if (!skp[j]) {
      float* op = out + (size_t)(b*NSEL_ + j)*5;
      op[0] = 0.f; op[1] = 0.f; op[2] = 0.f; op[3] = 0.f; op[4] = 0.f;
    }
  }
}

// ---------------------------------------------------------------------------
extern "C" void kernel_launch(void* const* d_in, const int* in_sizes, int n_in,
                              void* d_out, int out_size, void* d_ws, size_t ws_size,
                              hipStream_t stream) {
  (void)in_sizes; (void)n_in; (void)out_size; (void)ws_size;
  const float* x      = (const float*)d_in[0];
  const float* conv_w = (const float*)d_in[1];
  const float* conv_b = (const float*)d_in[2];
  const float* cls_w  = (const float*)d_in[3];
  const float* cls_b  = (const float*)d_in[4];
  const float* reg_w  = (const float*)d_in[5];
  const float* reg_b  = (const float*)d_in[6];
  float* out = (float*)d_out;

  char* ws = (char*)d_ws;
  size_t off = 0;
  auto alloc = [&](size_t bytes) -> void* {
    void* p = ws + off;
    off += (bytes + 255) & ~(size_t)255;
    return p;
  };
  uint16_t* xp    = (uint16_t*)alloc((size_t)B_*HP_*WP_*C_ * 2);       // 63.3 MB padded
  uint16_t* feat  = (uint16_t*)alloc((size_t)NPOS_ * C_ * 2);          // 62.3 MB
  uint16_t* wpack = (uint16_t*)alloc((size_t)9*16*8*32*16 * 2);        // 1.18 MB
  uint16_t* hwp   = (uint16_t*)alloc((size_t)8*32*16 * 2);
  float*    hbias = (float*)alloc(16 * 4);
  float*    hout  = (float*)alloc((size_t)NPOS_ * 16 * 4);             // 7.8 MB
  unsigned* hist  = (unsigned*)alloc((size_t)(2*NBUCKET_ + 4) * 4);
  int*      thrB  = (int*)(hist + 2*NBUCKET_);
  unsigned* cnt   = (unsigned*)(hist + 2*NBUCKET_ + 2);
  unsigned long long* cand = (unsigned long long*)alloc((size_t)B_*CANDCAP_*8);
  float*    topkL = (float*)alloc((size_t)B_*NSEL_ * 4);
  int*      topkI = (int*)alloc((size_t)B_*NSEL_ * 4);
  float*    boxes = (float*)alloc((size_t)B_*NSEL_*4 * 4);

  k_init      <<<33, 256, 0, stream>>>(hist);
  k_convert_x <<<(B_*32*H_*W_)/256, 256, 0, stream>>>(x, xp);          // 15200 blocks
  k_halo      <<<(B_*1012*32 + 255)/256, 256, 0, stream>>>(xp);
  k_pack_w    <<<2304, 256, 0, stream>>>(conv_w, wpack);
  k_pack_hw   <<<16, 256, 0, stream>>>(cls_w, reg_w, hwp);
  k_pack_hbias<<<1, 16, 0, stream>>>(cls_b, reg_b, hbias);
  k_conv      <<<7600, 256, 0, stream>>>(xp, wpack, conv_b, feat);     // 60800 waves x 144 WMMA
  k_head      <<<950, 256, 0, stream>>>(feat, hwp, hbias, hout);       // 7600 waves x 8 WMMA
  k_hist      <<<1425, 256, 0, stream>>>(hout, hist);
  k_thresh    <<<2, 32, 0, stream>>>(hist, thrB);
  k_collect   <<<1425, 256, 0, stream>>>(hout, thrB, cnt, cand);
  k_sort      <<<2, 1024, 0, stream>>>(cand, cnt, topkL, topkI);
  k_decode    <<<16, 256, 0, stream>>>(hout, topkL, topkI, boxes, out);
  k_nms       <<<2, 1024, 0, stream>>>(boxes, out);
}
